// MaskedMultiHeadAttention_8306466751124
// MI455X (gfx1250) — matrix operand order repaired and checked by reading, not compiled, not run on hardware
//
#include <hip/hip_runtime.h>

typedef _Float16 v16h __attribute__((ext_vector_type(16)));
typedef _Float16 v8h  __attribute__((ext_vector_type(8)));
typedef float    v8f  __attribute__((ext_vector_type(8)));
typedef float    v4f  __attribute__((ext_vector_type(4)));

#define N_HEADS  16
#define HEAD_DIM 64
#define EMBED    1024
#define BATCH    2
#define SEQ      2048
#define SRCLEN   128

union uh { v16h v; v8h h[2]; };

// Async global->LDS copy of 16 bytes (CDNA5 GLOBAL_LOAD_ASYNC_TO_LDS_B128).
// GV addressing: vaddr = full 64-bit VGPR pair, saddr = off.
// Generic pointer to __shared__ carries the LDS byte offset in its low 32 bits
__device__ __forceinline__ void async_copy_b128(const _Float16* gsrc, _Float16* ldst) {
    unsigned loff = (unsigned)(size_t)ldst;
    asm volatile("global_load_async_to_lds_b128 %0, %1, off"
                 :: "v"(loff), "v"(gsrc) : "memory");
}
__device__ __forceinline__ void wait_async_all() {
    asm volatile("s_wait_asynccnt 0" ::: "memory");
}

// ---------------------------------------------------------------------------
// Kernel 1: Y = X @ W^T + b  (M=B*T=4096, N=K=1024), output f16 in [B,H,T,Dh]
// blockDim=256 (8 waves), tile 128(M) x 64(N), K-step 32.
// grid: (M/128=32, N/64=16, 3)  z selects Wq/Wk/Wv
// ---------------------------------------------------------------------------
__global__ __launch_bounds__(256) void qkv_gemm(
    const float* __restrict__ X,
    const float* __restrict__ Wq, const float* __restrict__ bq,
    const float* __restrict__ Wk, const float* __restrict__ bk,
    const float* __restrict__ Wv, const float* __restrict__ bv,
    _Float16* __restrict__ qkv)
{
    __shared__ __align__(16) _Float16 Xs[128 * 48];   // row stride 48 halves (96B)
    __shared__ __align__(16) _Float16 Ws[64 * 48];

    const int z  = blockIdx.z;
    const float* W    = (z == 0) ? Wq : (z == 1) ? Wk : Wv;
    const float* bias = (z == 0) ? bq : (z == 1) ? bk : bv;
    _Float16* Y = qkv + (size_t)z * ((size_t)BATCH * SEQ * EMBED);

    const int m0 = blockIdx.x * 128;
    const int n0 = blockIdx.y * 64;
    const int tid  = threadIdx.x;
    const int lane = tid & 31;
    const int wave = tid >> 5;
    const int wr = wave >> 1;      // 0..3 -> M
    const int wc = wave & 1;       // 0..1 -> N
    const int l15 = lane & 15;
    const int hi  = lane >> 4;     // 0/1

    v8f acc[2][2] = {};

    for (int kk = 0; kk < EMBED; kk += 32) {
        __syncthreads();
        // ---- load X tile 128x32 (fp32 -> f16) : thread covers 16 floats ----
        {
            int row = tid >> 1;
            int cb  = (tid & 1) * 16;
            const float* src = X + (size_t)(m0 + row) * EMBED + kk + cb;
            _Float16* dst = &Xs[row * 48 + cb];
            #pragma unroll
            for (int i = 0; i < 2; ++i) {
                v4f f0 = *(const v4f*)(src + i * 8 + 0);
                v4f f1 = *(const v4f*)(src + i * 8 + 4);
                v8h hh;
                #pragma unroll
                for (int j = 0; j < 4; ++j) { hh[j] = (_Float16)f0[j]; hh[4 + j] = (_Float16)f1[j]; }
                *(v8h*)(dst + i * 8) = hh;
            }
        }
        // ---- load W tile 64x32 : thread covers 8 floats ----
        {
            int row = tid >> 2;
            int cb  = (tid & 3) * 8;
            const float* src = W + (size_t)(n0 + row) * EMBED + kk + cb;
            v4f f0 = *(const v4f*)(src + 0);
            v4f f1 = *(const v4f*)(src + 4);
            v8h hh;
            #pragma unroll
            for (int j = 0; j < 4; ++j) { hh[j] = (_Float16)f0[j]; hh[4 + j] = (_Float16)f1[j]; }
            *(v8h*)&Ws[row * 48 + cb] = hh;
        }
        __syncthreads();

        // ---- fragments + WMMA ----
        uh Bf[2];
        #pragma unroll
        for (int nt = 0; nt < 2; ++nt) {
            int n = wc * 32 + nt * 16 + l15;
            // B 32x16: lane n, elems 0..7 -> k = hi*8 + e ; elems 8..15 -> k = 16 + hi*8 + (e-8)
            Bf[nt].h[0] = *(const v8h*)&Ws[n * 48 + hi * 8];
            Bf[nt].h[1] = *(const v8h*)&Ws[n * 48 + hi * 8 + 16];
        }
        #pragma unroll
        for (int mt = 0; mt < 2; ++mt) {
            int m = wr * 32 + mt * 16 + l15;
            int kb = hi * 8;
            // A 16x32: lane m, elems 0..7 -> k=kb+0..7 ; elems 8..15 -> k=kb+16..23
            uh A;
            A.h[0] = *(const v8h*)&Xs[m * 48 + kb];
            A.h[1] = *(const v8h*)&Xs[m * 48 + kb + 16];
            #pragma unroll
            for (int nt = 0; nt < 2; ++nt) {
                acc[mt][nt] = __builtin_amdgcn_wmma_f32_16x16x32_f16(
                    false, A.v, false, Bf[nt].v, (short)0, acc[mt][nt], false, false);
            }
        }
    }

    // ---- epilogue: +bias, f16 store into [B,H,T,Dh] ----
    #pragma unroll
    for (int mt = 0; mt < 2; ++mt) {
        #pragma unroll
        for (int nt = 0; nt < 2; ++nt) {
            int n  = n0 + wc * 32 + nt * 16 + l15;
            float bv2 = bias[n];
            int h  = n >> 6;
            int dh = n & 63;
            #pragma unroll
            for (int r = 0; r < 8; ++r) {
                int m = m0 + wr * 32 + mt * 16 + r + hi * 8;
                int b = m >> 11;          // /SEQ
                int t = m & (SEQ - 1);
                size_t idx = ((((size_t)b * N_HEADS + h) * SEQ + t) * HEAD_DIM + dh);
                Y[idx] = (_Float16)(acc[mt][nt][r] + bv2);
            }
        }
    }
}

// ---------------------------------------------------------------------------
// Kernel 2: fused flash attention. grid (T/128=16, B*H=32), blockDim=256.
// Each wave owns 16 query rows; KV tiles of 64 keys.
// Q and K tiles use CDNA5 async global->LDS copies (ASYNCcnt); V is loaded
// through VGPRs because it is transposed into LDS.
// ---------------------------------------------------------------------------
__global__ __launch_bounds__(256) void attn_fwd(
    const _Float16* __restrict__ Qh,
    const _Float16* __restrict__ Kh,
    const _Float16* __restrict__ Vh,
    float* __restrict__ Out)
{
    __shared__ __align__(16) _Float16 Qs[128 * 72];     // [row][d]  stride 72 (144B)
    __shared__ __align__(16) _Float16 Ks[64 * 72];      // [j][d]
    __shared__ __align__(16) _Float16 Vt[64 * 72];      // [d][j]  (transposed)
    __shared__ __align__(16) _Float16 Ps[8 * 16 * 72];  // per-wave P strip [16][72]

    const int bh = blockIdx.y;             // b*16 + h
    const int i0 = blockIdx.x * 128;
    const int tid  = threadIdx.x;
    const int lane = tid & 31;
    const int wave = tid >> 5;
    const int l15 = lane & 15;
    const int hi  = lane >> 4;

    const size_t hd = (size_t)bh * SEQ * HEAD_DIM;
    const _Float16* Qg = Qh + hd;
    const _Float16* Kg = Kh + hd;
    const _Float16* Vg = Vh + hd;

    // ---- async-load Q tile 128x64 into LDS ----
    {
        int row = tid >> 1;
        int db  = (tid & 1) * 32;
        const _Float16* src = Qg + (size_t)(i0 + row) * HEAD_DIM + db;
        _Float16* dst = &Qs[row * 72 + db];
        #pragma unroll
        for (int i = 0; i < 4; ++i)
            async_copy_b128(src + i * 8, dst + i * 8);
    }
    wait_async_all();
    __syncthreads();

    // ---- Q fragments in registers (2 chunks over d) ----
    uh Aq[2];
    {
        int m  = wave * 16 + l15;
        int kb = hi * 8;
        #pragma unroll
        for (int c = 0; c < 2; ++c) {
            Aq[c].h[0] = *(const v8h*)&Qs[m * 72 + c * 32 + kb];
            Aq[c].h[1] = *(const v8h*)&Qs[m * 72 + c * 32 + kb + 16];
        }
    }

    v8f Oacc[4] = {};
    float Mrow[8], Lrow[8];
    #pragma unroll
    for (int r = 0; r < 8; ++r) { Mrow[r] = -1e30f; Lrow[r] = 0.0f; }

    const int iwave = i0 + wave * 16;
    const int jend  = min((int)SEQ, i0 + 128 + SRCLEN);
    const float scale = 0.125f;               // 1/sqrt(64)
    const float LOG2E = 1.4426950408889634f;
    _Float16* Pw = &Ps[wave * 16 * 72];

    for (int j0 = 0; j0 < jend; j0 += 64) {
        __syncthreads();
        // ---- async-load K tile 64x64 (global -> LDS, no VGPR round-trip) ----
        {
            int row = tid >> 2;
            int db  = (tid & 3) * 16;
            const _Float16* src = Kg + (size_t)(j0 + row) * HEAD_DIM + db;
            _Float16* dst = &Ks[row * 72 + db];
            async_copy_b128(src, dst);
            async_copy_b128(src + 8, dst + 8);
        }
        // ---- load V tile 64x64, transposed into Vt[d][j] ----
        {
            int row = tid >> 2;            // local j
            int db  = (tid & 3) * 16;      // d base
            const _Float16* src = Vg + (size_t)(j0 + row) * HEAD_DIM + db;
            v8h a = *(const v8h*)src;
            v8h b = *(const v8h*)(src + 8);
            #pragma unroll
            for (int e = 0; e < 8; ++e) {
                Vt[(db + e) * 72 + row]     = a[e];
                Vt[(db + 8 + e) * 72 + row] = b[e];
            }
        }
        // prefetch next tile's K/V lines into cache while we compute
        if (j0 + 64 < jend) {
            int row = tid >> 2;
            int db  = (tid & 3) * 16;
            __builtin_prefetch(Kg + (size_t)(j0 + 64 + row) * HEAD_DIM + db, 0, 3);
            __builtin_prefetch(Vg + (size_t)(j0 + 64 + row) * HEAD_DIM + db, 0, 3);
        }
        wait_async_all();     // K tile resident in LDS (per-wave portion)
        __syncthreads();

        // ---- S = Q K^T : 16 rows x 64 cols per wave ----
        v8f S[4] = {};
        #pragma unroll
        for (int nt = 0; nt < 4; ++nt) {
            int j = nt * 16 + l15;
            #pragma unroll
            for (int c = 0; c < 2; ++c) {
                uh Bk;
                Bk.h[0] = *(const v8h*)&Ks[j * 72 + c * 32 + hi * 8];
                Bk.h[1] = *(const v8h*)&Ks[j * 72 + c * 32 + hi * 8 + 16];
                S[nt] = __builtin_amdgcn_wmma_f32_16x16x32_f16(
                    false, Aq[c].v, false, Bk.v, (short)0, S[nt], false, false);
            }
        }

        // ---- scale + prefix-causal mask ----
        #pragma unroll
        for (int nt = 0; nt < 4; ++nt) {
            int j = j0 + nt * 16 + l15;
            #pragma unroll
            for (int r = 0; r < 8; ++r) {
                int i = iwave + r + hi * 8;
                float s = S[nt][r] * scale;
                S[nt][r] = (j <= i + SRCLEN) ? s : -1e30f;
            }
        }

        // ---- online softmax (per-row, 16-lane butterfly within half-wave) ----
        float alpha[8];
        #pragma unroll
        for (int r = 0; r < 8; ++r) {
            float mx = fmaxf(fmaxf(S[0][r], S[1][r]), fmaxf(S[2][r], S[3][r]));
            mx = fmaxf(mx, __shfl_xor(mx, 1, 32));
            mx = fmaxf(mx, __shfl_xor(mx, 2, 32));
            mx = fmaxf(mx, __shfl_xor(mx, 4, 32));
            mx = fmaxf(mx, __shfl_xor(mx, 8, 32));
            float newM = fmaxf(Mrow[r], mx);
            alpha[r] = exp2f((Mrow[r] - newM) * LOG2E);
            Mrow[r] = newM;
            float rs = 0.0f;
            #pragma unroll
            for (int nt = 0; nt < 4; ++nt) {
                float p = exp2f((S[nt][r] - newM) * LOG2E);
                S[nt][r] = p;
                rs += p;
            }
            rs += __shfl_xor(rs, 1, 32);
            rs += __shfl_xor(rs, 2, 32);
            rs += __shfl_xor(rs, 4, 32);
            rs += __shfl_xor(rs, 8, 32);
            Lrow[r] = Lrow[r] * alpha[r] + rs;
        }

        // ---- P -> LDS (f16), rescale O ----
        #pragma unroll
        for (int nt = 0; nt < 4; ++nt) {
            #pragma unroll
            for (int r = 0; r < 8; ++r) {
                Pw[(r + hi * 8) * 72 + nt * 16 + l15] = (_Float16)S[nt][r];
                Oacc[nt][r] *= alpha[r];
            }
        }
        // per-wave LDS strip: no __syncthreads needed, compiler orders via DScnt

        // ---- O += P V ----
        #pragma unroll
        for (int c = 0; c < 2; ++c) {          // key-dim chunk of 32
            uh Ap;
            int m  = l15;
            int kb = hi * 8;
            Ap.h[0] = *(const v8h*)&Pw[m * 72 + c * 32 + kb];
            Ap.h[1] = *(const v8h*)&Pw[m * 72 + c * 32 + kb + 16];
            #pragma unroll
            for (int nt = 0; nt < 4; ++nt) {
                int d = nt * 16 + l15;
                uh Bv;
                Bv.h[0] = *(const v8h*)&Vt[d * 72 + c * 32 + hi * 8];
                Bv.h[1] = *(const v8h*)&Vt[d * 72 + c * 32 + hi * 8 + 16];
                Oacc[nt] = __builtin_amdgcn_wmma_f32_16x16x32_f16(
                    false, Ap.v, false, Bv.v, (short)0, Oacc[nt], false, false);
            }
        }
    }

    // ---- epilogue: O / L, store fp32 to [B, T, C] ----
    const int b = bh >> 4;
    const int h = bh & 15;
    #pragma unroll
    for (int nt = 0; nt < 4; ++nt) {
        int d = nt * 16 + l15;
        #pragma unroll
        for (int r = 0; r < 8; ++r) {
            int t = iwave + r + hi * 8;
            float o = Oacc[nt][r] / Lrow[r];
            Out[((size_t)b * SEQ + t) * EMBED + h * HEAD_DIM + d] = o;
        }
    }
}

// ---------------------------------------------------------------------------
extern "C" void kernel_launch(void* const* d_in, const int* in_sizes, int n_in,
                              void* d_out, int out_size, void* d_ws, size_t ws_size,
                              hipStream_t stream) {
    const float* x  = (const float*)d_in[0];
    const float* Wq = (const float*)d_in[1];
    const float* bq = (const float*)d_in[2];
    const float* Wk = (const float*)d_in[3];
    const float* bk = (const float*)d_in[4];
    const float* Wv = (const float*)d_in[5];
    const float* bv = (const float*)d_in[6];

    _Float16* qkv = (_Float16*)d_ws;                 // q | k | v, each B*T*C f16
    const size_t QSZ = (size_t)BATCH * SEQ * EMBED;

    dim3 g1(32, 16, 3), b1(256);
    qkv_gemm<<<g1, b1, 0, stream>>>(x, Wq, bq, Wk, bk, Wv, bv, qkv);

    dim3 g2(SEQ / 128, BATCH * N_HEADS), b2(256);
    attn_fwd<<<g2, b2, 0, stream>>>(qkv, qkv + QSZ, qkv + 2 * QSZ, (float*)d_out);
}
